// DAFNetBaseLayer_88476326297689
// MI455X (gfx1250) — matrix operand order repaired and checked by reading, not compiled, not run on hardware
//
#include <hip/hip_runtime.h>

typedef __attribute__((ext_vector_type(16))) __bf16 v16bf;
typedef __attribute__((ext_vector_type(8)))  float  v8f;
typedef int v4i __attribute__((ext_vector_type(4)));

#define THREADS 256
#define KT 32          // K-step per stage (one WMMA depth)
#define BSTRIDE 40     // padded LDS row stride (elements) to spread banks

#if defined(__has_builtin)
#if __has_builtin(__builtin_amdgcn_global_load_async_to_lds_b128)
#define HAVE_ASYNC 1
#endif
#endif

#if defined(HAVE_ASYNC)
typedef __attribute__((address_space(1))) v4i gv4i;
typedef __attribute__((address_space(3))) v4i lv4i;
#define ASYNC_CP_B128(g, l) \
    __builtin_amdgcn_global_load_async_to_lds_b128((gv4i*)(g), (lv4i*)(l), 0, 0)
#if __has_builtin(__builtin_amdgcn_s_wait_asynccnt)
#define WAIT_ASYNC(n) __builtin_amdgcn_s_wait_asynccnt(n)
#else
#define WAIT_ASYNC(n) asm volatile("s_wait_asynccnt %0" :: "n"(n) : "memory")
#endif
#endif

// ---------------------------------------------------------------------------
// WMMA GEMM: C[M,Nc] = act(A[M,K] @ W[K,Nc] + bias)
// A: bf16 row-major [M,K]; Wt: bf16 [Nc,K] (pre-transposed weight);
// bias: fp32 [Nc] (nullable). relu: apply max(0,.). outbf16: store bf16 else f32.
// Block = 256 threads = 8 waves; block tile 256x64; each wave does 32 rows x 64 cols.
// B tile (64 cols x 32 K = 4KB) staged in LDS (async double-buffered), shared by
// all 8 waves; fragments read back with ds_load_b128.
// ---------------------------------------------------------------------------
__global__ __launch_bounds__(THREADS)
void gemm_wmma_kernel(const __bf16* __restrict__ A, const __bf16* __restrict__ Wt,
                      const float* __restrict__ bias, void* __restrict__ Cout,
                      int M, int Nc, int K, int relu, int outbf16)
{
    __shared__ __bf16 Bs[2][64 * BSTRIDE];

    const int tid  = threadIdx.x;
    const int wave = tid >> 5;
    const int lane = tid & 31;
    const int lrow = lane & 15;
    const int lhi  = lane >> 4;
    const int m0 = blockIdx.x * 256 + wave * 32;
    const int n0 = blockIdx.y * 64;

    // A row pointers for the wave's two 16-row tiles (clamped; OOB rows discarded at store)
    int ar0 = m0 + lrow;      if (ar0 >= M) ar0 = M - 1; if (ar0 < 0) ar0 = 0;
    int ar1 = m0 + 16 + lrow; if (ar1 >= M) ar1 = M - 1;
    const __bf16* Ap0 = A + (size_t)ar0 * K + lhi * 8;
    const __bf16* Ap1 = A + (size_t)ar1 * K + lhi * 8;

    // B staging assignment: thread -> (column, 8-element chunk); 256 threads cover 64x32 tile
    const int scol = tid >> 2;           // 0..63
    const int schk = tid & 3;            // 0..3
    int gcol = n0 + scol; if (gcol >= Nc) gcol = Nc - 1;
    const __bf16* Bg  = Wt + (size_t)gcol * K + schk * 8;
    __bf16* Bl[2] = { &Bs[0][scol * BSTRIDE + schk * 8],
                      &Bs[1][scol * BSTRIDE + schk * 8] };

    v8f acc[2][4] = {{v8f{}, v8f{}, v8f{}, v8f{}}, {v8f{}, v8f{}, v8f{}, v8f{}}};

    const int S = K / KT;   // number of K stages (K is a multiple of 32 here)

    // stage 0
#if defined(HAVE_ASYNC)
    ASYNC_CP_B128(Bg, Bl[0]);
#else
    *(uint4*)Bl[0] = *(const uint4*)Bg;
#endif

    for (int i = 0; i < S; ++i) {
        // issue next stage into the other buffer
        if (i + 1 < S) {
            const __bf16* g = Bg + (size_t)(i + 1) * KT;
#if defined(HAVE_ASYNC)
            ASYNC_CP_B128(g, Bl[(i + 1) & 1]);
#else
            *(uint4*)Bl[(i + 1) & 1] = *(const uint4*)g;
#endif
        }
#if defined(HAVE_ASYNC)
        if (i + 1 < S) { WAIT_ASYNC(1); } else { WAIT_ASYNC(0); }
#endif
        __syncthreads();   // stage i visible to all waves; prior compute done

        const int k0 = i * KT;
        __builtin_prefetch((const void*)(Ap0 + k0 + 64), 0, 1);
        __builtin_prefetch((const void*)(Ap1 + k0 + 64), 0, 1);

        // A fragments (direct global, per-wave rows)
        union { v16bf v; uint4 q[2]; } a0, a1;
        a0.q[0] = *(const uint4*)(Ap0 + k0);
        a0.q[1] = *(const uint4*)(Ap0 + k0 + 16);
        a1.q[0] = *(const uint4*)(Ap1 + k0);
        a1.q[1] = *(const uint4*)(Ap1 + k0 + 16);

        // B fragments from LDS: col = t*16+lrow, K chunks at lhi*8 and 16+lhi*8
        const __bf16* bufp = &Bs[i & 1][0];
        union { v16bf v; uint4 q[2]; } b0, b1, b2, b3;
        {
            const __bf16* p0 = bufp + (0 * 16 + lrow) * BSTRIDE + lhi * 8;
            const __bf16* p1 = bufp + (1 * 16 + lrow) * BSTRIDE + lhi * 8;
            const __bf16* p2 = bufp + (2 * 16 + lrow) * BSTRIDE + lhi * 8;
            const __bf16* p3 = bufp + (3 * 16 + lrow) * BSTRIDE + lhi * 8;
            b0.q[0] = *(const uint4*)p0; b0.q[1] = *(const uint4*)(p0 + 16);
            b1.q[0] = *(const uint4*)p1; b1.q[1] = *(const uint4*)(p1 + 16);
            b2.q[0] = *(const uint4*)p2; b2.q[1] = *(const uint4*)(p2 + 16);
            b3.q[0] = *(const uint4*)p3; b3.q[1] = *(const uint4*)(p3 + 16);
        }

        acc[0][0] = __builtin_amdgcn_wmma_f32_16x16x32_bf16(false, a0.v, false, b0.v, (short)0, acc[0][0], false, false);
        acc[0][1] = __builtin_amdgcn_wmma_f32_16x16x32_bf16(false, a0.v, false, b1.v, (short)0, acc[0][1], false, false);
        acc[0][2] = __builtin_amdgcn_wmma_f32_16x16x32_bf16(false, a0.v, false, b2.v, (short)0, acc[0][2], false, false);
        acc[0][3] = __builtin_amdgcn_wmma_f32_16x16x32_bf16(false, a0.v, false, b3.v, (short)0, acc[0][3], false, false);
        acc[1][0] = __builtin_amdgcn_wmma_f32_16x16x32_bf16(false, a1.v, false, b0.v, (short)0, acc[1][0], false, false);
        acc[1][1] = __builtin_amdgcn_wmma_f32_16x16x32_bf16(false, a1.v, false, b1.v, (short)0, acc[1][1], false, false);
        acc[1][2] = __builtin_amdgcn_wmma_f32_16x16x32_bf16(false, a1.v, false, b2.v, (short)0, acc[1][2], false, false);
        acc[1][3] = __builtin_amdgcn_wmma_f32_16x16x32_bf16(false, a1.v, false, b3.v, (short)0, acc[1][3], false, false);

        __syncthreads();   // done reading buf[i&1] before stage i+2 overwrites it
    }

    #pragma unroll
    for (int mt = 0; mt < 2; ++mt) {
        #pragma unroll
        for (int t = 0; t < 4; ++t) {
            int n = n0 + t * 16 + lrow;
            if (n >= Nc) continue;
            float bv = bias ? bias[n] : 0.0f;
            #pragma unroll
            for (int r = 0; r < 8; ++r) {
                int m = m0 + mt * 16 + lhi * 8 + r;
                if (m >= M) continue;
                float v = acc[mt][t][r] + bv;
                if (relu) v = v > 0.0f ? v : 0.0f;
                size_t off = (size_t)m * Nc + n;
                if (outbf16) ((__bf16*)Cout)[off] = (__bf16)v;
                else         ((float*)Cout)[off]  = v;
            }
        }
    }
}

// ---------------------------------------------------------------------------
// Weight transpose + fp32->bf16: W[K,Nc] -> Wt[Nc,K] (LDS tiled)
// ---------------------------------------------------------------------------
__global__ __launch_bounds__(THREADS)
void transpose_bf16_kernel(const float* __restrict__ W, __bf16* __restrict__ Wt, int K, int Nc)
{
    __shared__ float tile[16][17];
    int tx = threadIdx.x & 15, ty = threadIdx.x >> 4;
    int kb = blockIdx.x * 16, nb = blockIdx.y * 16;
    int k = kb + ty, n = nb + tx;
    tile[ty][tx] = (k < K && n < Nc) ? W[(size_t)k * Nc + n] : 0.0f;
    __syncthreads();
    int nn = nb + ty, kk = kb + tx;
    if (nn < Nc && kk < K) Wt[(size_t)nn * K + kk] = (__bf16)tile[tx][ty];
}

__global__ void f2bf_kernel(const float* __restrict__ in, __bf16* __restrict__ out, size_t n)
{
    size_t i = (size_t)blockIdx.x * THREADS + threadIdx.x;
    if (i < n) out[i] = (__bf16)in[i];
}

// ---------------------------------------------------------------------------
// Segment bookkeeping
// ---------------------------------------------------------------------------
__global__ void segprep_kernel(const int* __restrict__ lens, int* __restrict__ segoff, int E)
{
    if (blockIdx.x == 0 && threadIdx.x == 0) {
        int acc = 0;
        for (int e = 0; e < E; ++e) { segoff[e] = acc; acc += lens[e]; }
        segoff[E] = acc;
    }
}

__global__ void segfill_kernel(const int* __restrict__ segoff, int* __restrict__ seg)
{
    int e = blockIdx.x;
    for (int i = segoff[e] + threadIdx.x; i < segoff[e + 1]; i += blockDim.x) seg[i] = e;
}

// s[i] = sum_k h2[i,k]*w[k] + b
__global__ __launch_bounds__(THREADS)
void svec_kernel(const float* __restrict__ h2, const float* __restrict__ w,
                 const float* __restrict__ b, float* __restrict__ s, int K)
{
    __shared__ float red[THREADS];
    int i = blockIdx.x, tid = threadIdx.x;
    const float* row = h2 + (size_t)i * K;
    float acc = 0.0f;
    for (int k = tid; k < K; k += THREADS) acc += row[k] * w[k];
    red[tid] = acc; __syncthreads();
    for (int st = 128; st > 0; st >>= 1) { if (tid < st) red[tid] += red[tid + st]; __syncthreads(); }
    if (tid == 0) s[i] = red[0] + b[0];
}

__global__ __launch_bounds__(THREADS)
void segsoftmax_kernel(const int* __restrict__ segoff, const float* __restrict__ s,
                       float* __restrict__ wgt)
{
    __shared__ float red[THREADS];
    int e = blockIdx.x, tid = threadIdx.x;
    int lo = segoff[e], hi = segoff[e + 1];
    float m = -3.0e38f;
    for (int i = lo + tid; i < hi; i += THREADS) m = fmaxf(m, s[i]);
    red[tid] = m; __syncthreads();
    for (int st = 128; st > 0; st >>= 1) { if (tid < st) red[tid] = fmaxf(red[tid], red[tid + st]); __syncthreads(); }
    m = red[0]; __syncthreads();
    float sum = 0.0f;
    for (int i = lo + tid; i < hi; i += THREADS) { float p = __expf(s[i] - m); wgt[i] = p; sum += p; }
    red[tid] = sum; __syncthreads();
    for (int st = 128; st > 0; st >>= 1) { if (tid < st) red[tid] += red[tid + st]; __syncthreads(); }
    float den = red[0];
    for (int i = lo + tid; i < hi; i += THREADS) wgt[i] /= den;
}

// weighted[i,:] = attx[i,:]*wgt[i]; srep[e,:] = segment sum of weighted
__global__ __launch_bounds__(THREADS)
void segweighted_kernel(const int* __restrict__ segoff, const float* __restrict__ attx,
                        const float* __restrict__ wgt, float* __restrict__ weighted,
                        float* __restrict__ srep, int D)
{
    int e = blockIdx.x;
    int lo = segoff[e], hi = segoff[e + 1];
    for (int d = threadIdx.x; d < D; d += THREADS) {
        float acc = 0.0f;
        for (int i = lo; i < hi; ++i) {
            float v = attx[(size_t)i * D + d] * wgt[i];
            weighted[(size_t)i * D + d] = v;
            acc += v;
        }
        srep[(size_t)e * D + d] = acc;
    }
}

// out = LN(a + b[idx]) * w + bias  (b nullable, idx nullable, outb optional bf16 copy)
__global__ __launch_bounds__(THREADS)
void ln_kernel(const float* __restrict__ a, const float* __restrict__ b,
               const int* __restrict__ idxb, const float* __restrict__ w,
               const float* __restrict__ bias, float* __restrict__ out,
               __bf16* __restrict__ outb, int D)
{
    __shared__ float red[THREADS];
    int i = blockIdx.x, tid = threadIdx.x;
    const float* arow = a + (size_t)i * D;
    const float* brow = b ? b + (size_t)(idxb ? idxb[i] : i) * D : nullptr;
    float loc[4];
    int cnt = 0;
    float sum = 0.0f;
    for (int d = tid; d < D; d += THREADS) {
        float v = arow[d] + (brow ? brow[d] : 0.0f);
        loc[cnt++] = v; sum += v;
    }
    red[tid] = sum; __syncthreads();
    for (int st = 128; st > 0; st >>= 1) { if (tid < st) red[tid] += red[tid + st]; __syncthreads(); }
    float mean = red[0] / (float)D; __syncthreads();
    float vs = 0.0f;
    for (int j = 0; j < cnt; ++j) { float dd = loc[j] - mean; vs += dd * dd; }
    red[tid] = vs; __syncthreads();
    for (int st = 128; st > 0; st >>= 1) { if (tid < st) red[tid] += red[tid + st]; __syncthreads(); }
    float inv = rsqrtf(red[0] / (float)D + 1e-5f);
    cnt = 0;
    for (int d = tid; d < D; d += THREADS) {
        float v = (loc[cnt++] - mean) * inv * w[d] + bias[d];
        out[(size_t)i * D + d] = v;
        if (outb) outb[(size_t)i * D + d] = (__bf16)v;
    }
}

// new_rep = concat(past, srep) -> fp32 (into d_out) + bf16 copy
__global__ void buildrep_kernel(const float* __restrict__ past, const float* __restrict__ srep,
                                float* __restrict__ newrep, __bf16* __restrict__ nrb, int P, int D)
{
    int row = blockIdx.x;
    const float* src = (row < P) ? past + (size_t)row * D : srep + (size_t)(row - P) * D;
    for (int d = threadIdx.x; d < D; d += blockDim.x) {
        float v = src[d];
        newrep[(size_t)row * D + d] = v;
        nrb[(size_t)row * D + d] = (__bf16)v;
    }
}

// scores + softmax (full probs): a[h,q,k]; mask NEG when k >= q+P+1
__global__ __launch_bounds__(THREADS)
void attn_scores_kernel(const float* __restrict__ q, const float* __restrict__ k,
                        float* __restrict__ a, int H, int DK, int E, int PE, int P, float scale)
{
    __shared__ float red[THREADS];
    __shared__ float qs[64];
    int qi = blockIdx.x, h = blockIdx.y, tid = threadIdx.x;
    const float* qrow = q + (size_t)qi * (H * DK) + h * DK;
    if (tid < DK) qs[tid] = qrow[tid];
    __syncthreads();
    float sc[4]; int nk = 0; float m = -3.0e38f;
    for (int kk = tid; kk < PE; kk += THREADS) {
        const float* krow = k + (size_t)kk * (H * DK) + h * DK;
        float d = 0.0f;
        for (int t = 0; t < DK; ++t) d += qs[t] * krow[t];
        d = d * scale + ((kk >= qi + P + 1) ? -999999.0f : 0.0f);
        sc[nk++] = d; m = fmaxf(m, d);
    }
    red[tid] = m; __syncthreads();
    for (int st = 128; st > 0; st >>= 1) { if (tid < st) red[tid] = fmaxf(red[tid], red[tid + st]); __syncthreads(); }
    m = red[0]; __syncthreads();
    float sum = 0.0f;
    for (int j = 0; j < nk; ++j) { sc[j] = __expf(sc[j] - m); sum += sc[j]; }
    red[tid] = sum; __syncthreads();
    for (int st = 128; st > 0; st >>= 1) { if (tid < st) red[tid] += red[tid + st]; __syncthreads(); }
    float den = red[0];
    float* arow = a + ((size_t)h * E + qi) * PE;
    nk = 0;
    for (int kk = tid; kk < PE; kk += THREADS) arow[kk] = sc[nk++] / den;
}

// o[q, h*DK+d] = sum_k a[h,q,k] * v[k, h*DK+d]
__global__ __launch_bounds__(64)
void attn_av_kernel(const float* __restrict__ a, const float* __restrict__ v,
                    float* __restrict__ o, int H, int DK, int E, int PE)
{
    int qi = blockIdx.x, h = blockIdx.y;
    const float* arow = a + ((size_t)h * E + qi) * PE;
    for (int d = threadIdx.x; d < DK; d += blockDim.x) {
        float acc = 0.0f;
        for (int kk = 0; kk < PE; ++kk) acc += arow[kk] * v[(size_t)kk * (H * DK) + h * DK + d];
        o[(size_t)qi * (H * DK) + h * DK + d] = acc;
    }
}

// score attention: only diag softmax value u4[h,q] = softmax(sc[q,:])[P+q]
__global__ __launch_bounds__(THREADS)
void score_diag_kernel(const float* __restrict__ q2, const float* __restrict__ k2,
                       float* __restrict__ u4, int HS, int DKS, int E, int PE, int P, float scale)
{
    __shared__ float red[THREADS];
    __shared__ float qs[64];
    int qi = blockIdx.x, h = blockIdx.y, tid = threadIdx.x;
    const float* qrow = q2 + (size_t)qi * (HS * DKS) + h * DKS;
    if (tid < DKS) qs[tid] = qrow[tid];
    __syncthreads();
    float sc[4]; int nk = 0; float m = -3.0e38f; float diag = 0.0f; int have = 0;
    for (int kk = tid; kk < PE; kk += THREADS) {
        const float* krow = k2 + (size_t)kk * (HS * DKS) + h * DKS;
        float d = 0.0f;
        for (int t = 0; t < DKS; ++t) d += qs[t] * krow[t];
        d = d * scale + ((kk >= qi + P + 1) ? -999999.0f : 0.0f);
        if (kk == P + qi) { diag = d; have = 1; }
        sc[nk++] = d; m = fmaxf(m, d);
    }
    red[tid] = m; __syncthreads();
    for (int st = 128; st > 0; st >>= 1) { if (tid < st) red[tid] = fmaxf(red[tid], red[tid + st]); __syncthreads(); }
    m = red[0]; __syncthreads();
    float sum = 0.0f;
    for (int j = 0; j < nk; ++j) sum += __expf(sc[j] - m);
    red[tid] = sum; __syncthreads();
    for (int st = 128; st > 0; st >>= 1) { if (tid < st) red[tid] += red[tid + st]; __syncthreads(); }
    float den = red[0];
    if (have) u4[(size_t)h * E + qi] = __expf(diag - m) / den;
}

// head-mean + reversed cumprod -> past_update_scale, update_scale
__global__ void update_scale_kernel(const float* __restrict__ u4, float* __restrict__ past_scale,
                                    float* __restrict__ us, int HS, int E)
{
    if (blockIdx.x != 0 || threadIdx.x != 0) return;
    float u[256];
    for (int e = 0; e < E; ++e) {
        float a = 0.0f;
        for (int h = 0; h < HS; ++h) a += u4[(size_t)h * E + e];
        u[e] = a / (float)HS;
    }
    us[E - 1] = u[E - 1];
    float pf = 1.0f;
    for (int i = E - 2; i >= 0; --i) {
        pf *= (1.0f - u[i + 1]);      // pf = prod_{j=i+1..E-1}(1-u[j])
        us[i] = u[i] * pf;
    }
    past_scale[0] = pf * (1.0f - u[0]);
}

// FiLM gate: out = relu(x*scale[mode]+shift[mode]) + x
__global__ __launch_bounds__(THREADS)
void film_kernel(const float* __restrict__ x3, const int* __restrict__ mode,
                 const float* __restrict__ mscale, const float* __restrict__ mshift,
                 float* __restrict__ out, int D)
{
    int i = blockIdx.x;
    int m = mode[i];
    for (int d = threadIdx.x; d < D; d += THREADS) {
        float xv = x3[(size_t)i * D + d];
        float pre = xv * mscale[(size_t)m * D + d] + mshift[(size_t)m * D + d];
        out[(size_t)i * D + d] = (pre > 0.0f ? pre : 0.0f) + xv;
    }
}

// ---------------------------------------------------------------------------
extern "C" void kernel_launch(void* const* d_in, const int* in_sizes, int n_in,
                              void* d_out, int out_size, void* d_ws, size_t ws_size,
                              hipStream_t stream)
{
    const int D = 512, A_ = 512, H = 8, DK = 64, HS = 4, DKS = 64, FF = 2048, P = 512, E = 256;
    const int N = in_sizes[0] / D;
    const int PE = P + E;

    const float* x        = (const float*)d_in[0];
    const float* past_rep = (const float*)d_in[1];
    const int*   lens     = (const int*)  d_in[2];
    const int*   mode     = (const int*)  d_in[3];
    const float* t1w1 = (const float*)d_in[4];  const float* t1b1 = (const float*)d_in[5];
    const float* t1w2 = (const float*)d_in[6];  const float* t1b2 = (const float*)d_in[7];
    const float* t2w1 = (const float*)d_in[8];  const float* t2b1 = (const float*)d_in[9];
    const float* t2w2 = (const float*)d_in[10]; const float* t2b2 = (const float*)d_in[11];
    const float* oq_w = (const float*)d_in[12]; const float* oq_b = (const float*)d_in[13];
    const float* ok_w = (const float*)d_in[14]; const float* ok_b = (const float*)d_in[15];
    const float* ov_w = (const float*)d_in[16]; const float* ov_b = (const float*)d_in[17];
    const float* oo_w = (const float*)d_in[18]; const float* oo_b = (const float*)d_in[19];
    const float* sq_w = (const float*)d_in[20]; const float* sq_b = (const float*)d_in[21];
    const float* sk_w = (const float*)d_in[22]; const float* sk_b = (const float*)d_in[23];
    const float* f1w  = (const float*)d_in[24]; const float* f1b  = (const float*)d_in[25];
    const float* f2w  = (const float*)d_in[26]; const float* f2b  = (const float*)d_in[27];
    const float* ln1w = (const float*)d_in[28]; const float* ln1b = (const float*)d_in[29];
    const float* ln2w = (const float*)d_in[30]; const float* ln2b = (const float*)d_in[31];
    const float* ln3w = (const float*)d_in[32]; const float* ln3b = (const float*)d_in[33];
    const float* mscale = (const float*)d_in[34]; const float* mshift = (const float*)d_in[35];

    // d_out regions: [x: N*D][past_update_scale: 1][update_scale: E][new_rep: PE*D]
    float* out_x    = (float*)d_out;
    float* out_ps   = out_x + (size_t)N * D;
    float* out_us   = out_ps + 1;
    float* out_nrep = out_us + E;

    // workspace bump allocator
    char* ws = (char*)d_ws;
    size_t off = 0;
    auto alloc = [&](size_t bytes) -> void* {
        void* p = ws + off;
        off += (bytes + 255) & ~(size_t)255;
        return p;
    };

    __bf16* t1w1t = (__bf16*)alloc((size_t)D * A_ * 2);
    __bf16* t1w2t = (__bf16*)alloc((size_t)A_ * D * 2);
    __bf16* t2w1t = (__bf16*)alloc((size_t)D * A_ * 2);
    __bf16* oqt   = (__bf16*)alloc((size_t)D * (H * DK) * 2);
    __bf16* okt   = (__bf16*)alloc((size_t)D * (H * DK) * 2);
    __bf16* ovt   = (__bf16*)alloc((size_t)D * (H * DK) * 2);
    __bf16* oot   = (__bf16*)alloc((size_t)(H * DK) * D * 2);
    __bf16* sqt   = (__bf16*)alloc((size_t)D * (HS * DKS) * 2);
    __bf16* skt   = (__bf16*)alloc((size_t)D * (HS * DKS) * 2);
    __bf16* f1wt  = (__bf16*)alloc((size_t)D * FF * 2);
    __bf16* f2wt  = (__bf16*)alloc((size_t)FF * D * 2);

    __bf16* bufA  = (__bf16*)alloc((size_t)N * D * 2);   // xb -> attb -> x2b
    __bf16* bufB  = (__bf16*)alloc((size_t)N * D * 2);   // h1b
    float*  bufF1 = (float*) alloc((size_t)N * D * 4);   // att_x -> x1 -> ffn
    float*  bufF2 = (float*) alloc((size_t)N * D * 4);   // h2 -> weighted -> x2 -> x3
    __bf16* bufFF = (__bf16*)alloc((size_t)N * FF * 2);  // FFN hidden (bf16)
    float*  svecb = (float*) alloc((size_t)N * 4);
    float*  wgt   = (float*) alloc((size_t)N * 4);
    int*    seg   = (int*)   alloc((size_t)N * 4);
    int*    segoff= (int*)   alloc((size_t)(E + 1) * 4);
    float*  srep  = (float*) alloc((size_t)E * D * 4);
    __bf16* nrb   = (__bf16*)alloc((size_t)PE * D * 2);
    float*  qbuf  = (float*) alloc((size_t)E * (H * DK) * 4);
    float*  kbuf  = (float*) alloc((size_t)PE * (H * DK) * 4);
    float*  vbuf  = (float*) alloc((size_t)PE * (H * DK) * 4);
    float*  abuf  = (float*) alloc((size_t)H * E * PE * 4);
    float*  obuf  = (float*) alloc((size_t)E * (H * DK) * 4);
    __bf16* obb   = (__bf16*)alloc((size_t)E * (H * DK) * 2);
    float*  oproj = (float*) alloc((size_t)E * D * 4);
    float*  q2buf = (float*) alloc((size_t)E * (HS * DKS) * 4);
    float*  k2buf = (float*) alloc((size_t)PE * (HS * DKS) * 4);
    float*  u4    = (float*) alloc((size_t)HS * E * 4);

    auto gemm = [&](const __bf16* Am, const __bf16* Wt, const float* bias, void* C,
                    int M, int Nc, int K, int relu, int obf) {
        dim3 g((M + 255) / 256, (Nc + 63) / 64);
        gemm_wmma_kernel<<<g, THREADS, 0, stream>>>(Am, Wt, bias, C, M, Nc, K, relu, obf);
    };
    auto tpose = [&](const float* W, __bf16* Wt, int K, int Nc) {
        dim3 g(K / 16, Nc / 16);
        transpose_bf16_kernel<<<g, THREADS, 0, stream>>>(W, Wt, K, Nc);
    };

    // segments
    segprep_kernel<<<1, 32, 0, stream>>>(lens, segoff, E);
    segfill_kernel<<<E, THREADS, 0, stream>>>(segoff, seg);

    // weights -> bf16 transposed
    tpose(t1w1, t1w1t, D, A_);
    tpose(t1w2, t1w2t, A_, D);
    tpose(t2w1, t2w1t, D, A_);
    tpose(oq_w, oqt, D, H * DK);
    tpose(ok_w, okt, D, H * DK);
    tpose(ov_w, ovt, D, H * DK);
    tpose(oo_w, oot, H * DK, D);
    tpose(sq_w, sqt, D, HS * DKS);
    tpose(sk_w, skt, D, HS * DKS);
    tpose(f1w, f1wt, D, FF);
    tpose(f2w, f2wt, FF, D);

    // ---- inner attention ----
    size_t nx = (size_t)N * D;
    f2bf_kernel<<<(unsigned)((nx + THREADS - 1) / THREADS), THREADS, 0, stream>>>(x, bufA, nx);
    gemm(bufA, t1w1t, t1b1, bufB, N, A_, D, 1, 1);          // h1b = relu(x@t1w1+b), bf16
    gemm(bufB, t1w2t, t1b2, bufF1, N, D, A_, 0, 0);         // att_x = h1@t1w2+b, f32
    f2bf_kernel<<<(unsigned)((nx + THREADS - 1) / THREADS), THREADS, 0, stream>>>(bufF1, bufA, nx);
    gemm(bufA, t2w1t, t2b1, bufF2, N, A_, D, 1, 0);         // h2 = relu(att_x@t2w1+b), f32
    svec_kernel<<<N, THREADS, 0, stream>>>(bufF2, t2w2, t2b2, svecb, A_);
    segsoftmax_kernel<<<E, THREADS, 0, stream>>>(segoff, svecb, wgt);
    segweighted_kernel<<<E, THREADS, 0, stream>>>(segoff, bufF1, wgt, bufF2, srep, D);
    ln_kernel<<<N, THREADS, 0, stream>>>(x, bufF2, nullptr, ln1w, ln1b, bufF1, nullptr, D); // x1

    // ---- outer attention ----
    buildrep_kernel<<<PE, THREADS, 0, stream>>>(past_rep, srep, out_nrep, nrb, P, D);
    const __bf16* srb = nrb + (size_t)P * D;
    gemm(srb, oqt, oq_b, qbuf, E, H * DK, D, 0, 0);
    gemm(nrb, okt, ok_b, kbuf, PE, H * DK, D, 0, 0);
    gemm(nrb, ovt, ov_b, vbuf, PE, H * DK, D, 0, 0);
    {
        dim3 g(E, H);
        attn_scores_kernel<<<g, THREADS, 0, stream>>>(qbuf, kbuf, abuf, H, DK, E, PE, P, 0.125f);
        attn_av_kernel<<<g, 64, 0, stream>>>(abuf, vbuf, obuf, H, DK, E, PE);
    }
    size_t no = (size_t)E * H * DK;
    f2bf_kernel<<<(unsigned)((no + THREADS - 1) / THREADS), THREADS, 0, stream>>>(obuf, obb, no);
    gemm(obb, oot, oo_b, oproj, E, D, H * DK, 0, 0);
    ln_kernel<<<N, THREADS, 0, stream>>>(bufF1, oproj, seg, ln2w, ln2b, bufF2, bufA, D); // x2 (+bf16)

    // ---- outer score attention ----
    gemm(srb, sqt, sq_b, q2buf, E, HS * DKS, D, 0, 0);
    gemm(nrb, skt, sk_b, k2buf, PE, HS * DKS, D, 0, 0);
    {
        dim3 g(E, HS);
        score_diag_kernel<<<g, THREADS, 0, stream>>>(q2buf, k2buf, u4, HS, DKS, E, PE, P, 0.125f);
    }
    update_scale_kernel<<<1, 32, 0, stream>>>(u4, out_ps, out_us, HS, E);

    // ---- FFN + FiLM ----
    gemm(bufA, f1wt, f1b, bufFF, N, FF, D, 1, 1);           // f1 = relu(x2@f1w+b), bf16
    gemm(bufFF, f2wt, f2b, bufF1, N, D, FF, 0, 0);          // ffn, f32
    ln_kernel<<<N, THREADS, 0, stream>>>(bufF2, bufF1, nullptr, ln3w, ln3b, bufF2, nullptr, D); // x3
    film_kernel<<<N, THREADS, 0, stream>>>(bufF2, mode, mscale, mshift, out_x, D);
}
